// GRFExactAttention_13889924235743
// MI455X (gfx1250) — matrix operand order repaired and checked by reading, not compiled, not run on hardware
//
#include <hip/hip_runtime.h>
#include <hip/hip_bf16.h>

typedef __bf16 bf16;
typedef __attribute__((ext_vector_type(16))) __bf16 v16bf;
typedef __attribute__((ext_vector_type(8)))  __bf16 bf16x8;
typedef __attribute__((ext_vector_type(8)))  float  v8f;
typedef __attribute__((ext_vector_type(4)))  float  f32x4;

union AFragU { v16bf v; bf16x8 h[2]; };

// ---------------------------------------------------------------------------
// Kernel 1: qkv = x @ w_qkv   (M=8192, N=1152, K=384), f32 in -> bf16 frags
// Epilogue: relu+eps on q,k; scatter to Qb/Kb [BH][N][64], Vt [BH][64][N]
// ---------------------------------------------------------------------------
__global__ __launch_bounds__(256) void k_qkv(const float* __restrict__ x,
                                             const float* __restrict__ w,
                                             bf16* __restrict__ Qb,
                                             bf16* __restrict__ Kb,
                                             bf16* __restrict__ Vt) {
  const int lane = threadIdx.x & 31;
  const int wave = threadIdx.x >> 5;
  const int ln = lane & 15, hh = lane >> 4;
  const int mw = wave & 3, nw = wave >> 2;              // 4x2 wave grid
  const int rowBase = blockIdx.x * 128 + mw * 32;       // + mi*16
  const int colBase = blockIdx.y * 64 + nw * 32;        // + ni*16

  v8f acc[2][2] = {};
  for (int kc = 0; kc < 384; kc += 32) {
    AFragU a[2];
#pragma unroll
    for (int mi = 0; mi < 2; ++mi) {
      const float* xp = x + (size_t)(rowBase + mi * 16 + ln) * 384 + kc;
      // e<8: k=8h+e ; e>=8: k=16+8h+(e-8)
      f32x4 f0 = *(const f32x4*)(xp + 8 * hh);
      f32x4 f1 = *(const f32x4*)(xp + 8 * hh + 4);
      f32x4 f2 = *(const f32x4*)(xp + 16 + 8 * hh);
      f32x4 f3 = *(const f32x4*)(xp + 16 + 8 * hh + 4);
#pragma unroll
      for (int e = 0; e < 4; ++e) {
        a[mi].h[0][e]     = (bf16)f0[e];
        a[mi].h[0][e + 4] = (bf16)f1[e];
        a[mi].h[1][e]     = (bf16)f2[e];
        a[mi].h[1][e + 4] = (bf16)f3[e];
      }
    }
    v16bf b[2];
#pragma unroll
    for (int ni = 0; ni < 2; ++ni) {
      // B elem e: kk = 8h + e (e<8), 16 + 8h + (e-8) (e>=8), col = colBase + ni*16 + ln (coalesced over ln)
      const float* wp = w + (size_t)(kc + 8 * hh) * 1152 + (colBase + ni * 16 + ln);
#pragma unroll
      for (int e = 0; e < 16; ++e) b[ni][e] = (bf16)wp[(size_t)(e + (e & 8)) * 1152];
    }
#pragma unroll
    for (int mi = 0; mi < 2; ++mi)
#pragma unroll
      for (int ni = 0; ni < 2; ++ni)
        acc[mi][ni] = __builtin_amdgcn_wmma_f32_16x16x32_bf16(
            false, a[mi].v, false, b[ni], (short)0, acc[mi][ni], false, false);
  }
  // Epilogue: C/D layout row = r + 8h, col = ln
#pragma unroll
  for (int mi = 0; mi < 2; ++mi)
#pragma unroll
    for (int ni = 0; ni < 2; ++ni)
#pragma unroll
      for (int r = 0; r < 8; ++r) {
        int gr = rowBase + mi * 16 + r + 8 * hh;
        int gc = colBase + ni * 16 + ln;
        float v = acc[mi][ni][r];
        int bi = gr >> 10, n = gr & 1023;
        int sec = gc / 384, c = gc % 384;
        int head = c >> 6, d = c & 63;
        size_t bh = (size_t)bi * 6 + head;
        if (sec == 0) {
          Qb[(bh * 1024 + n) * 64 + d] = (bf16)(fmaxf(v, 0.f) + 1e-6f);
        } else if (sec == 1) {
          Kb[(bh * 1024 + n) * 64 + d] = (bf16)(fmaxf(v, 0.f) + 1e-6f);
        } else {
          Vt[(bh * 64 + d) * 1024 + n] = (bf16)v;   // transposed for B-frags
        }
      }
}

// ---------------------------------------------------------------------------
// Kernel 2: fused masked linear attention per (b,h). Wave owns 16 queries.
// S = (Q Kt) * mask ; rowsum via WMMA vs all-ones B ; O = S V ; O *= 1/(rs+eps)
// ---------------------------------------------------------------------------
__global__ __launch_bounds__(256) void k_attn(const bf16* __restrict__ Qb,
                                              const bf16* __restrict__ Kb,
                                              const bf16* __restrict__ Vt,
                                              const float* __restrict__ mask,
                                              bf16* __restrict__ attn) {
  __shared__ bf16 sS[8][16][32];   // per-wave 16x32 bf16 scratch (D->A relayout)
  const int lane = threadIdx.x & 31;
  const int wave = threadIdx.x >> 5;
  const int ln = lane & 15, hh = lane >> 4;
  const int bh = blockIdx.x;                       // 0..47
  const int qb = blockIdx.y * 128 + wave * 16;     // query tile base
  const int b = bh / 6, head = bh % 6;

  // Q A-fragments for the two 32-wide D chunks (loaded once)
  AFragU aQ[2];
  {
    const bf16* qp = Qb + ((size_t)bh * 1024 + (qb + ln)) * 64;
#pragma unroll
    for (int di = 0; di < 2; ++di) {
      aQ[di].h[0] = *(const bf16x8*)(qp + 32 * di + 16 * hh);
      aQ[di].h[1] = *(const bf16x8*)(qp + 32 * di + 16 * hh + 8);
    }
  }
  v16bf ones;
#pragma unroll
  for (int e = 0; e < 16; ++e) ones[e] = (bf16)1.0f;

  v8f O[4] = {};
  v8f rs = {};

  for (int kb = 0; kb < 1024; kb += 32) {
#pragma unroll
    for (int half = 0; half < 2; ++half) {
      const int key = kb + 16 * half + ln;         // B column = key
      const bf16* kp = Kb + ((size_t)bh * 1024 + key) * 64;
      v16bf bK0 = *(const v16bf*)(kp + 16 * hh);        // d = e + 16h
      v16bf bK1 = *(const v16bf*)(kp + 32 + 16 * hh);   // d = 32 + e + 16h
      v8f S = {};
      S = __builtin_amdgcn_wmma_f32_16x16x32_bf16(false, aQ[0].v, false, bK0,
                                                  (short)0, S, false, false);
      S = __builtin_amdgcn_wmma_f32_16x16x32_bf16(false, aQ[1].v, false, bK1,
                                                  (short)0, S, false, false);
      const float* mp = mask + (size_t)(qb + 8 * hh) * 1024 + key;
#pragma unroll
      for (int r = 0; r < 8; ++r) {
        float s = S[r] * mp[(size_t)r * 1024];     // row = qb + r + 8h
        sS[wave][r + 8 * hh][16 * half + ln] = (bf16)s;
      }
    }
    // Re-load masked S as an A fragment (16 rows x 32 keys); wave-private LDS,
    // same-wave DS ops are in order, compiler inserts the dscnt wait.
    AFragU aS;
    aS.h[0] = *(const bf16x8*)&sS[wave][ln][16 * hh];
    aS.h[1] = *(const bf16x8*)&sS[wave][ln][16 * hh + 8];
#pragma unroll
    for (int j = 0; j < 4; ++j) {
      // B elem e: key = kb + e + 16h, col = d = j*16 + ln  (Vt contiguous)
      const bf16* vp = Vt + ((size_t)bh * 64 + j * 16 + ln) * 1024 + kb + 16 * hh;
      v16bf bV = *(const v16bf*)vp;
      O[j] = __builtin_amdgcn_wmma_f32_16x16x32_bf16(false, aS.v, false, bV,
                                                     (short)0, O[j], false, false);
    }
    rs = __builtin_amdgcn_wmma_f32_16x16x32_bf16(false, aS.v, false, ones,
                                                 (short)0, rs, false, false);
  }

  // Epilogue: O layout matches rs layout (row r+8h, col ln) -> pure per-lane scale
#pragma unroll
  for (int r = 0; r < 8; ++r) {
    float z = 1.0f / (rs[r] + 1e-6f);
    int n = qb + r + 8 * hh;
    size_t base = ((size_t)b * 1024 + n) * 384 + head * 64;
#pragma unroll
    for (int j = 0; j < 4; ++j)
      attn[base + j * 16 + ln] = (bf16)(O[j][r] * z);
  }
}

// ---------------------------------------------------------------------------
// Kernel 3: out = attn @ w_out + b_out   (M=8192, N=384, K=384) -> f32
// ---------------------------------------------------------------------------
__global__ __launch_bounds__(256) void k_out(const bf16* __restrict__ attn,
                                             const float* __restrict__ w,
                                             const float* __restrict__ bias,
                                             float* __restrict__ out) {
  const int lane = threadIdx.x & 31;
  const int wave = threadIdx.x >> 5;
  const int ln = lane & 15, hh = lane >> 4;
  const int mw = wave & 3, nw = wave >> 2;
  const int rowBase = blockIdx.x * 128 + mw * 32;
  const int colBase = blockIdx.y * 64 + nw * 32;

  v8f acc[2][2] = {};
  for (int kc = 0; kc < 384; kc += 32) {
    AFragU a[2];
#pragma unroll
    for (int mi = 0; mi < 2; ++mi) {
      const bf16* ap = attn + (size_t)(rowBase + mi * 16 + ln) * 384 + kc;
      a[mi].h[0] = *(const bf16x8*)(ap + 8 * hh);
      a[mi].h[1] = *(const bf16x8*)(ap + 16 + 8 * hh);
    }
    v16bf b[2];
#pragma unroll
    for (int ni = 0; ni < 2; ++ni) {
      const float* wp = w + (size_t)(kc + 8 * hh) * 384 + (colBase + ni * 16 + ln);
#pragma unroll
      for (int e = 0; e < 16; ++e) b[ni][e] = (bf16)wp[(size_t)(e + (e & 8)) * 384];
    }
#pragma unroll
    for (int mi = 0; mi < 2; ++mi)
#pragma unroll
      for (int ni = 0; ni < 2; ++ni)
        acc[mi][ni] = __builtin_amdgcn_wmma_f32_16x16x32_bf16(
            false, a[mi].v, false, b[ni], (short)0, acc[mi][ni], false, false);
  }
#pragma unroll
  for (int mi = 0; mi < 2; ++mi)
#pragma unroll
    for (int ni = 0; ni < 2; ++ni)
#pragma unroll
      for (int r = 0; r < 8; ++r) {
        int gr = rowBase + mi * 16 + r + 8 * hh;
        int gc = colBase + ni * 16 + ln;
        out[(size_t)gr * 384 + gc] = acc[mi][ni][r] + bias[gc];
      }
}

// ---------------------------------------------------------------------------
extern "C" void kernel_launch(void* const* d_in, const int* in_sizes, int n_in,
                              void* d_out, int out_size, void* d_ws, size_t ws_size,
                              hipStream_t stream) {
  const float* x     = (const float*)d_in[0];   // [8,1024,384]
  const float* w_qkv = (const float*)d_in[1];   // [384,1152]
  const float* w_out = (const float*)d_in[2];   // [384,384]
  const float* b_out = (const float*)d_in[3];   // [384]
  const float* mask  = (const float*)d_in[4];   // [1024,1024]
  float* out = (float*)d_out;                   // [8,1024,384]

  const size_t szQ = (size_t)8 * 6 * 1024 * 64; // elements per bf16 buffer
  bf16* Qb   = (bf16*)d_ws;
  bf16* Kb   = Qb + szQ;
  bf16* Vt   = Kb + szQ;
  bf16* attn = Vt + szQ;                        // total 4 * 6.29 MB = 25.2 MB

  k_qkv<<<dim3(64, 18), 256, 0, stream>>>(x, w_qkv, Qb, Kb, Vt);
  k_attn<<<dim3(48, 8), 256, 0, stream>>>(Qb, Kb, Vt, mask, attn);
  k_out<<<dim3(64, 6), 256, 0, stream>>>(attn, w_out, b_out, out);
}
